// ResidualBlock_88021059764276
// MI455X (gfx1250) — matrix operand order repaired and checked by reading, not compiled, not run on hardware
//
#include <hip/hip_runtime.h>
#include <cstdint>
#include <cstddef>

// ---------------------------------------------------------------- constants
#define S_LEN   8192
#define D_MODEL 768
#define D_RNN   1024
#define MLP_IN  3072
#define EPS     1e-5f

#define NCHUNK  64
#define CHUNK   128   // NCHUNK * CHUNK == S_LEN

// ---------------------------------------------------------------- types
typedef __attribute__((ext_vector_type(16))) __bf16 v16bf;
typedef __attribute__((ext_vector_type(8)))  __bf16 v8bf;
typedef __attribute__((ext_vector_type(8)))  float  v8f;

union FragAB { v16bf v; v8bf h[2]; };

enum { GF_BIAS = 1, GF_ACCUM = 2, GF_RESID = 4 };

// ---------------------------------------------------------------- helpers
__device__ __forceinline__ float gelu_tanh(float x) {
  const float k0 = 0.7978845608028654f, k1 = 0.044715f;
  float x3 = x * x * x;
  return 0.5f * x * (1.0f + tanhf(k0 * (x + k1 * x3)));
}
__device__ __forceinline__ float sigm(float x) { return 1.0f / (1.0f + expf(-x)); }

// CDNA5 async global->LDS copy, 16B per lane, tracked by ASYNCcnt.
// vdst operand = VGPR holding LDS byte address (low 32 bits of generic
// shared-aperture address, per ISA 10.2).
__device__ __forceinline__ void async_copy_b128(uint32_t lds_off, const void* gaddr) {
  asm volatile("global_load_async_to_lds_b128 %0, %1, off"
               :: "v"(lds_off), "v"(gaddr) : "memory");
}
__device__ __forceinline__ void wait_asynccnt0() {
#if __has_builtin(__builtin_amdgcn_s_wait_asynccnt)
  __builtin_amdgcn_s_wait_asynccnt(0);
#else
  asm volatile("s_wait_asynccnt 0x0" ::: "memory");
#endif
}
__device__ __forceinline__ uint32_t lds_off32(const void* p) {
  return (uint32_t)(uintptr_t)p;
}

// ================================================================
// WMMA bf16 GEMM:  C[M,N] = A[M,K] * B[N,K]^T  (+bias[N]) (+C) (+src[M,N])
// Block tile 128x128x32, 256 threads = 8 waves, wave tile 64x32 (4x2 WMMA).
// Tiles staged with GLOBAL_LOAD_ASYNC_TO_LDS_B128, double buffered.
// ================================================================
__global__ __launch_bounds__(256)
void gemm_bf16_wmma(const __bf16* __restrict__ A, const __bf16* __restrict__ B,
                    float* __restrict__ C, const float* __restrict__ bias,
                    const float* __restrict__ src,
                    int M, int N, int K, int flags)
{
  constexpr int BM = 128, BN = 128, BK = 32, PAD = 8;
  __shared__ alignas(16) __bf16 As[2][BM][BK + PAD];
  __shared__ alignas(16) __bf16 Bs[2][BN][BK + PAD];

  const int tid  = threadIdx.x;
  const int lane = tid & 31;
  const int wave = tid >> 5;
  const int wm   = (wave & 1) * 64;   // wave M offset in block tile
  const int wn   = (wave >> 1) * 32;  // wave N offset in block tile
  const int row0 = blockIdx.y * BM;
  const int col0 = blockIdx.x * BN;

  v8f acc[4][2];
#pragma unroll
  for (int mi = 0; mi < 4; ++mi)
#pragma unroll
    for (int ni = 0; ni < 2; ++ni)
      acc[mi][ni] = (v8f){0.f,0.f,0.f,0.f,0.f,0.f,0.f,0.f};

  const int r  = tid >> 2;        // 0..63, staging row
  const int cg = (tid & 3) * 8;   // staging col group (8 bf16 = 16B)

  // stage one 128x32 A tile + 128x32 B tile into LDS buffer `buf` (async)
  auto stage = [&](int buf, int k0) {
    const __bf16* ga = A + (size_t)(row0 + r) * K + k0 + cg;
    async_copy_b128(lds_off32(&As[buf][r][cg]),      ga);
    async_copy_b128(lds_off32(&As[buf][r + 64][cg]), ga + (size_t)64 * K);
    const __bf16* gb = B + (size_t)(col0 + r) * K + k0 + cg;
    async_copy_b128(lds_off32(&Bs[buf][r][cg]),      gb);
    async_copy_b128(lds_off32(&Bs[buf][r + 64][cg]), gb + (size_t)64 * K);
  };

  const int nk = K / BK;
  stage(0, 0);

  for (int ki = 0; ki < nk; ++ki) {
    const int cur = ki & 1;
    // all of this wave's async copies into `cur` complete, then rendezvous
    wait_asynccnt0();
    __syncthreads();
    // prefetch next tile into the other buffer (its readers passed the barrier)
    if (ki + 1 < nk) stage(1 - cur, (ki + 1) * BK);

    // ---- load fragments per ISA layouts (CDNA5 ISA §7.12.2)
    FragAB fa[4], fb[2];
    const int ml = lane & 15;
    const int kh = (lane >> 4) * 8;    // A: lanes 16-31 start at K=8 (then +16)
#pragma unroll
    for (int mi = 0; mi < 4; ++mi) {
      const __bf16* p = &As[cur][wm + mi * 16 + ml][kh];
      fa[mi].h[0] = *(const v8bf*)p;          // K = kh .. kh+7
      fa[mi].h[1] = *(const v8bf*)(p + 16);   // K = kh+16 .. kh+23
    }
    const int kq = (lane >> 4) * 8;    // B: lanes 16-31 start at K=8 (then +16)
#pragma unroll
    for (int ni = 0; ni < 2; ++ni) {
      const __bf16* p = &Bs[cur][wn + ni * 16 + ml][kq];
      fb[ni].h[0] = *(const v8bf*)p;          // K = kq .. kq+7
      fb[ni].h[1] = *(const v8bf*)(p + 16);   // K = kq+16 .. kq+23
    }

#pragma unroll
    for (int mi = 0; mi < 4; ++mi)
#pragma unroll
      for (int ni = 0; ni < 2; ++ni)
        acc[mi][ni] = __builtin_amdgcn_wmma_f32_16x16x32_bf16(
            false, fa[mi].v, false, fb[ni].v, (short)0, acc[mi][ni],
            false, false);
  }

  // ---- epilogue: C/D layout — lane gives N, VGPR index + lane/16 gives M
  const int ml = lane & 15;
  const int mh = (lane >> 4) * 8;
#pragma unroll
  for (int mi = 0; mi < 4; ++mi)
#pragma unroll
    for (int ni = 0; ni < 2; ++ni) {
      const int col = col0 + wn + ni * 16 + ml;
      float bv = (flags & GF_BIAS) ? bias[col] : 0.f;
#pragma unroll
      for (int v = 0; v < 8; ++v) {
        const size_t idx = (size_t)(row0 + wm + mi * 16 + mh + v) * N + col;
        float val = acc[mi][ni][v] + bv;
        if (flags & GF_ACCUM) val += C[idx];
        if (flags & GF_RESID) val += src[idx];
        C[idx] = val;
      }
    }
}

// ================================================================ elementwise
__global__ void rmsnorm_bf16_kernel(const float* __restrict__ x,
                                    const float* __restrict__ w,
                                    __bf16* __restrict__ out, int D)
{
  const int row = blockIdx.x;
  const float* xr = x + (size_t)row * D;
  float ss = 0.f;
  for (int c = threadIdx.x; c < D; c += blockDim.x) { float v = xr[c]; ss += v * v; }
#pragma unroll
  for (int o = 16; o > 0; o >>= 1) ss += __shfl_xor(ss, o, 32);
  __shared__ float red[8];
  const int lane = threadIdx.x & 31, wv = threadIdx.x >> 5;
  if (lane == 0) red[wv] = ss;
  __syncthreads();
  if (wv == 0) {
    float t = (lane < 8) ? red[lane] : 0.f;
#pragma unroll
    for (int o = 4; o > 0; o >>= 1) t += __shfl_xor(t, o, 32);
    if (lane == 0) red[0] = t;
  }
  __syncthreads();
  const float inv = rsqrtf(red[0] / (float)D + EPS);
  for (int c = threadIdx.x; c < D; c += blockDim.x)
    out[(size_t)row * D + c] = (__bf16)(xr[c] * inv * w[c]);
}

__global__ void f32_to_bf16_kernel(const float* __restrict__ in,
                                   __bf16* __restrict__ out, size_t n)
{
  size_t i = (size_t)blockIdx.x * blockDim.x + threadIdx.x;
  if (i < n) out[i] = (__bf16)in[i];
}

// w_conv (O, I, 4) -> 4 matrices [k][O, I] in bf16
__global__ void conv_w_rearrange_kernel(const float* __restrict__ w,
                                        __bf16* __restrict__ out)
{
  size_t i = (size_t)blockIdx.x * blockDim.x + threadIdx.x;
  const size_t n = (size_t)4 * D_RNN * D_RNN;
  if (i >= n) return;
  const size_t k  = i / ((size_t)D_RNN * D_RNN);
  const size_t oi = i % ((size_t)D_RNN * D_RNN);
  out[i] = (__bf16)w[oi * 4 + k];
}

__global__ void zero_bf16_kernel(__bf16* p, size_t n)
{
  size_t i = (size_t)blockIdx.x * blockDim.x + threadIdx.x;
  if (i < n) p[i] = (__bf16)0.f;
}

// a_branch = gelu(ab[:, :D_RNN]) in place; bbpad[t+3] = bf16(ab[:, D_RNN:])
__global__ void gelu_split_kernel(float* __restrict__ ab,
                                  __bf16* __restrict__ bbpad)
{
  size_t i = (size_t)blockIdx.x * blockDim.x + threadIdx.x;
  if (i >= (size_t)S_LEN * D_RNN) return;
  const size_t t = i / D_RNN;
  const int    c = (int)(i % D_RNN);
  float* rowp = ab + t * (size_t)(2 * D_RNN);
  rowp[c] = gelu_tanh(rowp[c]);
  bbpad[(t + 3) * D_RNN + c] = (__bf16)rowp[D_RNN + c];
}

// RG-LRU gates: a = exp(-8*softplus(Lambda)*rec_gate); gx = sqrt(1-a^2)*in_gate*conv
__global__ void gates_kernel(const float* __restrict__ pg,
                             const float* __restrict__ conv,
                             const float* __restrict__ Lambda,
                             float* __restrict__ aArr, float* __restrict__ gx)
{
  size_t i = (size_t)blockIdx.x * blockDim.x + threadIdx.x;
  if (i >= (size_t)S_LEN * D_RNN) return;
  const size_t t = i / D_RNN;
  const int    c = (int)(i % D_RNN);
  const float ig = sigm(pg[t * (size_t)(2 * D_RNN) + c]);
  const float rg = sigm(pg[t * (size_t)(2 * D_RNN) + D_RNN + c]);
  const float sp = log1pf(expf(Lambda[c]));
  const float av = expf(-8.0f * sp * rg);
  aArr[i] = av;
  gx[i]   = sqrtf(fmaxf(1.0f - av * av, 0.f)) * ig * conv[i];
}

// ---- chunked linear scan h_t = a_t h_{t-1} + g_t (3 passes) ----
__global__ void scan_chunk_kernel(const float* __restrict__ a,
                                  const float* __restrict__ gx,
                                  float* __restrict__ Pq, float* __restrict__ Sq)
{
  const int idx = blockIdx.x * blockDim.x + threadIdx.x;  // < NCHUNK*D_RNN
  const int c = idx & (D_RNN - 1);
  const int q = idx / D_RNN;
  float P = 1.f, Ssum = 0.f;
  size_t base = (size_t)q * CHUNK * D_RNN + c;
  for (int t = 0; t < CHUNK; ++t) {
    const float av = a[base], g = gx[base];
    P *= av;
    Ssum = av * Ssum + g;
    base += D_RNN;
  }
  Pq[(size_t)q * D_RNN + c] = P;
  Sq[(size_t)q * D_RNN + c] = Ssum;
}

__global__ void scan_carry_kernel(const float* __restrict__ Pq,
                                  const float* __restrict__ Sq,
                                  float* __restrict__ init)
{
  const int c = blockIdx.x * blockDim.x + threadIdx.x;  // < D_RNN
  float h = 0.f;
  for (int q = 0; q < NCHUNK; ++q) {
    init[(size_t)q * D_RNN + c] = h;
    h = Pq[(size_t)q * D_RNN + c] * h + Sq[(size_t)q * D_RNN + c];
  }
}

// apply pass, fused with ybf = bf16(a_branch * y); a_branch lives in ab stride 2*D_RNN
__global__ void scan_apply_kernel(const float* __restrict__ a,
                                  const float* __restrict__ gx,
                                  const float* __restrict__ init,
                                  const float* __restrict__ abr,
                                  __bf16* __restrict__ ybf)
{
  const int idx = blockIdx.x * blockDim.x + threadIdx.x;
  const int c = idx & (D_RNN - 1);
  const int q = idx / D_RNN;
  float h = init[(size_t)q * D_RNN + c];
  size_t base  = (size_t)q * CHUNK * D_RNN + c;
  size_t abase = (size_t)q * CHUNK * (size_t)(2 * D_RNN) + c;
  for (int t = 0; t < CHUNK; ++t) {
    h = a[base] * h + gx[base];
    ybf[base] = (__bf16)(abr[abase] * h);
    base  += D_RNN;
    abase += 2 * D_RNN;
  }
}

// gated MLP inner: mibf = bf16(gelu(m[:, :MLP_IN]) * m[:, MLP_IN:])
__global__ void mlp_gate_kernel(const float* __restrict__ m,
                                __bf16* __restrict__ mibf)
{
  size_t i = (size_t)blockIdx.x * blockDim.x + threadIdx.x;
  if (i >= (size_t)S_LEN * MLP_IN) return;
  const size_t t = i / MLP_IN;
  const int    j = (int)(i % MLP_IN);
  const float g = gelu_tanh(m[t * (size_t)(2 * MLP_IN) + j]);
  const float u = m[t * (size_t)(2 * MLP_IN) + MLP_IN + j];
  mibf[i] = (__bf16)(g * u);
}

// ================================================================ launcher
extern "C" void kernel_launch(void* const* d_in, const int* in_sizes, int n_in,
                              void* d_out, int out_size, void* d_ws, size_t ws_size,
                              hipStream_t stream)
{
  (void)in_sizes; (void)n_in; (void)out_size; (void)ws_size;
  const float* x      = (const float*)d_in[0];
  const float* w_n1   = (const float*)d_in[1];
  const float* w_n2   = (const float*)d_in[2];
  const float* w_lin1 = (const float*)d_in[3];
  const float* b_lin1 = (const float*)d_in[4];
  const float* w_conv = (const float*)d_in[5];
  const float* w_rg   = (const float*)d_in[6];
  const float* b_rg   = (const float*)d_in[7];
  const float* Lambda = (const float*)d_in[8];
  const float* w_out  = (const float*)d_in[9];
  const float* b_out  = (const float*)d_in[10];
  const float* w_m1   = (const float*)d_in[11];
  const float* b_m1   = (const float*)d_in[12];
  const float* w_m2   = (const float*)d_in[13];
  const float* b_m2   = (const float*)d_in[14];
  float* out = (float*)d_out;

  // ---------------- workspace layout (256B aligned carves) ----------------
  char* W = (char*)d_ws;
  size_t off = 0;
  auto take = [&](size_t b) { size_t r = off; off = (off + b + 255) & ~(size_t)255; return r; };

  __bf16* hbf     = (__bf16*)(W + take((size_t)S_LEN * D_MODEL * 2));
  __bf16* lin1bf  = (__bf16*)(W + take((size_t)2 * D_RNN * D_MODEL * 2));
  __bf16* rgbf    = (__bf16*)(W + take((size_t)2 * D_RNN * D_RNN * 2));
  __bf16* outwbf  = (__bf16*)(W + take((size_t)D_MODEL * D_RNN * 2));
  __bf16* m1bf    = (__bf16*)(W + take((size_t)2 * MLP_IN * D_MODEL * 2));
  __bf16* m2bf    = (__bf16*)(W + take((size_t)D_MODEL * MLP_IN * 2));
  __bf16* convwbf = (__bf16*)(W + take((size_t)4 * D_RNN * D_RNN * 2));

  // Big region: phase-1 {ab, bbpad, convf, convbf, pregates} aliased by phase-2 m
  const size_t sz_ab    = (size_t)S_LEN * 2 * D_RNN * 4;        // 67.1 MB
  const size_t sz_bbpad = (size_t)(S_LEN + 3) * D_RNN * 2;      // 16.8 MB
  const size_t sz_convf = (size_t)S_LEN * D_RNN * 4;            // 33.6 MB
  const size_t sz_cvbf  = (size_t)S_LEN * D_RNN * 2;            // 16.8 MB
  const size_t sz_pg    = (size_t)S_LEN * 2 * D_RNN * 4;        // 67.1 MB
  const size_t sz_m     = (size_t)S_LEN * 2 * MLP_IN * 4;       // 201.3 MB
  size_t phase1 = sz_ab + sz_bbpad + sz_convf + sz_cvbf + sz_pg;
  size_t bigOff = take(phase1 > sz_m ? phase1 : sz_m);
  float*  ab    = (float*) (W + bigOff);
  __bf16* bbpad = (__bf16*)(W + bigOff + sz_ab);
  float*  convf = (float*) (W + bigOff + sz_ab + sz_bbpad);
  __bf16* cvbf  = (__bf16*)(W + bigOff + sz_ab + sz_bbpad + sz_convf);
  float*  pg    = (float*) (W + bigOff + sz_ab + sz_bbpad + sz_convf + sz_cvbf);
  float*  mbuf  = (float*) (W + bigOff);                        // alias (phase 2)

  // Region 2: {a, gx} aliased by mibf (consumed before phase 2)
  size_t r2 = take(2 * ((size_t)S_LEN * D_RNN * 4 + 256));
  float*  aArr = (float*)(W + r2);
  float*  gxA  = (float*)(W + r2 + (size_t)S_LEN * D_RNN * 4 + 256);
  __bf16* mibf = (__bf16*)(W + r2);                              // alias (phase 2)

  __bf16* ybf  = (__bf16*)(W + take((size_t)S_LEN * D_RNN * 2));
  float*  x1   = (float*) (W + take((size_t)S_LEN * D_MODEL * 4));
  __bf16* h2bf = (__bf16*)(W + take((size_t)S_LEN * D_MODEL * 2));
  float*  Pq   = (float*) (W + take((size_t)NCHUNK * D_RNN * 4));
  float*  Sq   = (float*) (W + take((size_t)NCHUNK * D_RNN * 4));
  float*  init = (float*) (W + take((size_t)NCHUNK * D_RNN * 4));

  auto blk = [](size_t n) { return (unsigned)((n + 255) / 256); };
  auto gemm = [&](const __bf16* A, const __bf16* Bw, float* C,
                  const float* bias, const float* src, int M, int N, int K, int flags) {
    dim3 g((unsigned)(N / 128), (unsigned)(M / 128));
    gemm_bf16_wmma<<<g, 256, 0, stream>>>(A, Bw, C, bias, src, M, N, K, flags);
  };

  // ---------------- weight conversion (fp32 -> bf16) ----------------
  f32_to_bf16_kernel<<<blk((size_t)2*D_RNN*D_MODEL), 256, 0, stream>>>(w_lin1, lin1bf, (size_t)2*D_RNN*D_MODEL);
  f32_to_bf16_kernel<<<blk((size_t)2*D_RNN*D_RNN),   256, 0, stream>>>(w_rg,   rgbf,   (size_t)2*D_RNN*D_RNN);
  f32_to_bf16_kernel<<<blk((size_t)D_MODEL*D_RNN),   256, 0, stream>>>(w_out,  outwbf, (size_t)D_MODEL*D_RNN);
  f32_to_bf16_kernel<<<blk((size_t)2*MLP_IN*D_MODEL),256, 0, stream>>>(w_m1,   m1bf,   (size_t)2*MLP_IN*D_MODEL);
  f32_to_bf16_kernel<<<blk((size_t)D_MODEL*MLP_IN),  256, 0, stream>>>(w_m2,   m2bf,   (size_t)D_MODEL*MLP_IN);
  conv_w_rearrange_kernel<<<blk((size_t)4*D_RNN*D_RNN), 256, 0, stream>>>(w_conv, convwbf);

  // ---------------- forward ----------------
  // h = rms(x) * w_n1  (bf16)
  rmsnorm_bf16_kernel<<<S_LEN, 256, 0, stream>>>(x, w_n1, hbf, D_MODEL);
  // ab = h @ w_lin1^T + b_lin1
  gemm(hbf, lin1bf, ab, b_lin1, nullptr, S_LEN, 2 * D_RNN, D_MODEL, GF_BIAS);
  // a_branch = gelu(ab[:, :DR]) in place; bbpad = front-padded bf16 b_branch
  zero_bf16_kernel<<<blk((size_t)3 * D_RNN), 256, 0, stream>>>(bbpad, (size_t)3 * D_RNN);
  gelu_split_kernel<<<blk((size_t)S_LEN * D_RNN), 256, 0, stream>>>(ab, bbpad);
  // dense causal conv == 4 shifted accumulating GEMMs
  for (int k = 0; k < 4; ++k)
    gemm(bbpad + (size_t)k * D_RNN, convwbf + (size_t)k * D_RNN * D_RNN,
         convf, nullptr, nullptr, S_LEN, D_RNN, D_RNN, k == 0 ? 0 : GF_ACCUM);
  // gates = sigmoid(conv @ w_rg^T + b_rg)
  f32_to_bf16_kernel<<<blk((size_t)S_LEN * D_RNN), 256, 0, stream>>>(convf, cvbf, (size_t)S_LEN * D_RNN);
  gemm(cvbf, rgbf, pg, b_rg, nullptr, S_LEN, 2 * D_RNN, D_RNN, GF_BIAS);
  gates_kernel<<<blk((size_t)S_LEN * D_RNN), 256, 0, stream>>>(pg, convf, Lambda, aArr, gxA);
  // chunked linear scan + fused (a_branch * y) -> bf16
  scan_chunk_kernel<<<(NCHUNK * D_RNN) / 256, 256, 0, stream>>>(aArr, gxA, Pq, Sq);
  scan_carry_kernel<<<D_RNN / 256, 256, 0, stream>>>(Pq, Sq, init);
  scan_apply_kernel<<<(NCHUNK * D_RNN) / 256, 256, 0, stream>>>(aArr, gxA, init, ab, ybf);
  // x1 = (a_branch*y) @ w_out^T + b_out + x
  gemm(ybf, outwbf, x1, b_out, x, S_LEN, D_MODEL, D_RNN, GF_BIAS | GF_RESID);
  // h2 = rms(x1) * w_n2
  rmsnorm_bf16_kernel<<<S_LEN, 256, 0, stream>>>(x1, w_n2, h2bf, D_MODEL);
  // m = h2 @ w_m1^T + b_m1   (aliases phase-1 region, now dead)
  gemm(h2bf, m1bf, mbuf, b_m1, nullptr, S_LEN, 2 * MLP_IN, D_MODEL, GF_BIAS);
  // mi = gelu(m[:, :MI]) * m[:, MI:]
  mlp_gate_kernel<<<blk((size_t)S_LEN * MLP_IN), 256, 0, stream>>>(mbuf, mibf);
  // out = mi @ w_m2^T + b_m2 + x1
  gemm(mibf, m2bf, out, b_m2, x1, S_LEN, D_MODEL, MLP_IN, GF_BIAS | GF_RESID);
}
